// SelfAttention_54717883351429
// MI455X (gfx1250) — matrix operand order repaired and checked by reading, not compiled, not run on hardware
//
#include <hip/hip_runtime.h>
#include <hip/hip_bf16.h>

#define NSEQ   4096
#define DMODEL 1024
#define NHEAD  16
#define HD     64

typedef __attribute__((ext_vector_type(16))) _Float16 v16h;
typedef __attribute__((ext_vector_type(8)))  _Float16 v8h;
typedef __attribute__((ext_vector_type(4)))  _Float16 v4h;
typedef __attribute__((ext_vector_type(8)))  float    v8f;
typedef __attribute__((ext_vector_type(4)))  unsigned v4u;
typedef __attribute__((ext_vector_type(8)))  int      v8i_t;
typedef __attribute__((ext_vector_type(4)))  int      v4i_t;

// TDM available only as the 5-arg builtin on ROCm 7.2 (clang-22); the
// therock toolchain ships amd_gfx1250_TDM.h and a 6-arg builtin instead.
#if defined(__has_builtin)
#if !__has_include(<hip/amd_detail/amd_gfx1250_TDM.h>) && \
    __has_builtin(__builtin_amdgcn_tensor_load_to_lds)
#define USE_TDM 1
#endif
#endif
#ifndef USE_TDM
#define USE_TDM 0
#endif

__device__ __forceinline__ v8f wmma16(v16h a, v16h b, v8f c) {
  return __builtin_amdgcn_wmma_f32_16x16x32_f16(false, a, false, b, (short)0, c,
                                                false, false);
}

// Per-lane async global->LDS 16-byte copy (ASYNCcnt).
__device__ __forceinline__ void async_copy_b128(unsigned lds_off,
                                                const _Float16* g) {
  unsigned long long ga = (unsigned long long)(size_t)g;
  asm volatile("global_load_async_to_lds_b128 %0, %1, off"
               :: "v"(lds_off), "v"(ga) : "memory");
}

__device__ __forceinline__ void wait_asynccnt0() {
#if defined(__has_builtin) && __has_builtin(__builtin_amdgcn_s_wait_asynccnt)
  __builtin_amdgcn_s_wait_asynccnt(0);
#else
  asm volatile("s_wait_asynccnt 0x0" ::: "memory");
#endif
}

#if USE_TDM
// 2-D TDM tile load: `rows` rows of `row_halves` f16, row stride in halves.
__device__ __forceinline__ void tdm_load_2d(unsigned lds_addr, const void* gptr,
                                            unsigned rows, unsigned row_halves,
                                            unsigned stride_halves) {
  unsigned long long ga = (unsigned long long)(size_t)gptr;
  v4u g0;
  g0[0] = 1u;                                              // count=1
  g0[1] = lds_addr;                                        // lds byte addr
  g0[2] = (unsigned)ga;                                    // global[31:0]
  g0[3] = (unsigned)((ga >> 32) & 0x01ffffffu) | (2u << 30);  // [56:32]|type=2
  v8i_t g1;
  const unsigned td = 1u << 20;  // large tensor dims: tile always in bounds
  g1[0] = (int)(1u << 16);                 // data_size = 2 bytes
  g1[1] = (int)((td & 0xffffu) << 16);     // tensor_dim0[15:0]
  g1[2] = (int)((td >> 16) | ((td & 0xffffu) << 16));      // td0 hi | td1 lo
  g1[3] = (int)((td >> 16) | (row_halves << 16));          // td1 hi | tile0
  g1[4] = (int)rows;                       // tile_dim1 | tile_dim2=0
  g1[5] = (int)stride_halves;              // tensor_dim0_stride[31:0]
  g1[6] = 0;
  g1[7] = 0;
  v4i_t z = {0, 0, 0, 0};
  __builtin_amdgcn_tensor_load_to_lds(g0, g1, z, z, 0);
}
#endif

// ---------------------------------------------------------------- fp32->fp16
__global__ __launch_bounds__(256) void cvt_f32_f16(const float* __restrict__ s,
                                                   _Float16* __restrict__ d,
                                                   int n4) {
  int i = blockIdx.x * blockDim.x + threadIdx.x;
  if (i < n4) {
    float4 f = ((const float4*)s)[i];
    v4h h;
    h[0] = (_Float16)f.x; h[1] = (_Float16)f.y;
    h[2] = (_Float16)f.z; h[3] = (_Float16)f.w;
    ((v4h*)d)[i] = h;
  }
}

// ------------------------------------------------------- QKV projection GEMM
// out[n, o] = sum_c x[n, c] * W[o, c]; wave tile 32(M) x 64(N), block 128x128.
__global__ __launch_bounds__(256) void qkv_proj_kernel(
    const _Float16* __restrict__ xh,
    const _Float16* __restrict__ wqh, const _Float16* __restrict__ wkh,
    const _Float16* __restrict__ wvh,
    _Float16* __restrict__ qh, _Float16* __restrict__ kh,
    _Float16* __restrict__ vh) {
  const _Float16* W; _Float16* O;
  if (blockIdx.z == 0)      { W = wqh; O = qh; }
  else if (blockIdx.z == 1) { W = wkh; O = kh; }
  else                      { W = wvh; O = vh; }
  const int lane = threadIdx.x & 31;
  const int wave = threadIdx.x >> 5;
  const int hi   = lane >> 4;
  const int col  = lane & 15;
  const int m0 = blockIdx.x * 128 + (wave & 3) * 32;
  const int n0 = blockIdx.y * 128 + (wave >> 2) * 64;

  v8f acc[2][4] = {};
  const _Float16* arow0 = xh + (size_t)(m0 + col) * DMODEL + hi * 8;
  const _Float16* arow1 = arow0 + 16 * (size_t)DMODEL;
  for (int kk = 0; kk < DMODEL; kk += 32) {
    __builtin_prefetch(arow0 + kk + 128, 0, 0);
    v16h a[2];
    {
      v8h a0 = *(const v8h*)(arow0 + kk);
      v8h a1 = *(const v8h*)(arow0 + kk + 16);
      v8h a2 = *(const v8h*)(arow1 + kk);
      v8h a3 = *(const v8h*)(arow1 + kk + 16);
      #pragma unroll
      for (int i = 0; i < 8; ++i) {
        a[0][i] = a0[i]; a[0][8 + i] = a1[i];
        a[1][i] = a2[i]; a[1][8 + i] = a3[i];
      }
    }
    #pragma unroll
    for (int j = 0; j < 4; ++j) {
      const _Float16* brow =
          W + (size_t)(n0 + 16 * j + col) * DMODEL + kk + hi * 8;
      v8h b0 = *(const v8h*)brow;
      v8h b1 = *(const v8h*)(brow + 16);
      v16h b;
      #pragma unroll
      for (int i = 0; i < 8; ++i) { b[i] = b0[i]; b[8 + i] = b1[i]; }
      acc[0][j] = wmma16(a[0], b, acc[0][j]);
      acc[1][j] = wmma16(a[1], b, acc[1][j]);
    }
  }
  #pragma unroll
  for (int mi = 0; mi < 2; ++mi)
    #pragma unroll
    for (int j = 0; j < 4; ++j)
      #pragma unroll
      for (int r = 0; r < 8; ++r)
        O[(size_t)(m0 + 16 * mi + r + hi * 8) * DMODEL + n0 + 16 * j + col] =
            (_Float16)acc[mi][j][r];
}

// ------------------------------------------------------------ flash attention
__global__ __launch_bounds__(256) void flash_attn_kernel(
    const _Float16* __restrict__ qh, const _Float16* __restrict__ kh,
    const _Float16* __restrict__ vh, _Float16* __restrict__ ctxh) {
  __shared__ _Float16 Kt[32 * HD];      // 4 KB  (TDM-filled when available)
  __shared__ _Float16 Vt[32 * HD];      // 4 KB  (async-to-LDS filled)
  __shared__ float    Px[8][16 * 32];   // 16 KB per-wave P relayout scratch

  const int tid  = threadIdx.x;
  const int lane = tid & 31;
  const int wave = tid >> 5;
  const int hi   = lane >> 4;
  const int col  = lane & 15;
  const int head = blockIdx.y;
  const int m0   = blockIdx.x * 128 + wave * 16;

  // Preload Q A-tiles (16 x 64 = two K=32 chunks), resident in VGPRs
  v16h qa[2];
  {
    const _Float16* qrow =
        qh + (size_t)(m0 + col) * DMODEL + head * HD + hi * 8;
    #pragma unroll
    for (int c = 0; c < 2; ++c) {
      v8h a0 = *(const v8h*)(qrow + c * 32);
      v8h a1 = *(const v8h*)(qrow + c * 32 + 16);
      #pragma unroll
      for (int i = 0; i < 8; ++i) { qa[c][i] = a0[i]; qa[c][8 + i] = a1[i]; }
    }
  }

  v8f acc[4] = {};
  float mrow[8], lrow[8];
  #pragma unroll
  for (int r = 0; r < 8; ++r) { mrow[r] = -3.0e38f; lrow[r] = 0.0f; }

  const int key = tid >> 3;
  const int dd  = (tid & 7) * 8;

  for (int kb = 0; kb < NSEQ; kb += 32) {
    // ---- stage 32 keys + values into LDS via CDNA5 async paths ----
    {
      const size_t g = (size_t)(kb + key) * DMODEL + head * HD + dd;
      async_copy_b128((unsigned)(size_t)&Vt[key * HD + dd], vh + g);
#if !USE_TDM
      async_copy_b128((unsigned)(size_t)&Kt[key * HD + dd], kh + g);
#endif
    }
#if USE_TDM
    if (wave == 0) {
      tdm_load_2d((unsigned)(size_t)&Kt[0],
                  kh + (size_t)kb * DMODEL + head * HD, 32, HD, DMODEL);
      __builtin_amdgcn_s_wait_tensorcnt(0);
    }
#endif
    wait_asynccnt0();
    __syncthreads();

    // S(16x32) = Q @ K^T : two 16x16 tiles, each over two K=32 chunks
    v8f s[2];
    #pragma unroll
    for (int j = 0; j < 2; ++j) {
      v8f c = {};
      #pragma unroll
      for (int dc = 0; dc < 2; ++dc) {
        const _Float16* brow = &Kt[(j * 16 + col) * HD + dc * 32 + hi * 8];
        v8h b0 = *(const v8h*)brow;
        v8h b1 = *(const v8h*)(brow + 16);
        v16h b;
        #pragma unroll
        for (int i = 0; i < 8; ++i) { b[i] = b0[i]; b[8 + i] = b1[i]; }
        c = wmma16(qa[dc], b, c);
      }
      s[j] = c;
    }

    // Online softmax (each lane: 8 rows; 16-lane half-group col reduction)
    const float scale = 0.125f;  // 1/sqrt(64)
    float alpha[8];
    #pragma unroll
    for (int r = 0; r < 8; ++r) {
      float v0 = s[0][r] * scale;
      float v1 = s[1][r] * scale;
      float mx = fmaxf(v0, v1);
      #pragma unroll
      for (int off = 1; off < 16; off <<= 1)
        mx = fmaxf(mx, __shfl_xor(mx, off, 32));
      const float newm = fmaxf(mrow[r], mx);
      alpha[r] = __expf(mrow[r] - newm);
      const float p0 = __expf(v0 - newm);
      const float p1 = __expf(v1 - newm);
      float psum = p0 + p1;
      #pragma unroll
      for (int off = 1; off < 16; off <<= 1)
        psum += __shfl_xor(psum, off, 32);
      lrow[r] = lrow[r] * alpha[r] + psum;
      mrow[r] = newm;
      const int row = r + hi * 8;
      Px[wave][row * 32 + col]      = p0;
      Px[wave][row * 32 + 16 + col] = p1;
    }
    asm volatile("s_wait_dscnt 0x0" ::: "memory");  // wave-local LDS RAW fence

    // Re-read P as a 16x32 f16 A-tile
    v16h pa;
    {
      const float* prow = &Px[wave][col * 32 + hi * 8];
      #pragma unroll
      for (int i = 0; i < 8; ++i) {
        pa[i]     = (_Float16)prow[i];
        pa[8 + i] = (_Float16)prow[16 + i];
      }
    }

    // Rescale running context by alpha, accumulate P @ V (4 WMMAs)
    #pragma unroll
    for (int t = 0; t < 4; ++t) {
      v8f c = acc[t];
      #pragma unroll
      for (int r = 0; r < 8; ++r) c[r] *= alpha[r];
      v16h b;
      #pragma unroll
      for (int i = 0; i < 16; ++i)
        b[i] = Vt[((i & 8) * 2 + hi * 8 + (i & 7)) * HD + t * 16 + col];
      acc[t] = wmma16(pa, b, c);
    }
    __syncthreads();  // protect Kt/Vt before next stage
  }

  float rl[8];
  #pragma unroll
  for (int r = 0; r < 8; ++r) rl[r] = 1.0f / lrow[r];
  #pragma unroll
  for (int t = 0; t < 4; ++t)
    #pragma unroll
    for (int r = 0; r < 8; ++r)
      ctxh[(size_t)(m0 + r + hi * 8) * DMODEL + head * HD + t * 16 + col] =
          (_Float16)(acc[t][r] * rl[r]);
}

// -------------------------------------------------- output projection + bias
__global__ __launch_bounds__(256) void out_proj_kernel(
    const _Float16* __restrict__ ctxh, const _Float16* __restrict__ woh,
    const float* __restrict__ bo, float* __restrict__ out) {
  const int lane = threadIdx.x & 31;
  const int wave = threadIdx.x >> 5;
  const int hi   = lane >> 4;
  const int col  = lane & 15;
  const int m0 = blockIdx.x * 128 + (wave & 3) * 32;
  const int n0 = blockIdx.y * 128 + (wave >> 2) * 64;

  v8f acc[2][4] = {};
  const _Float16* arow0 = ctxh + (size_t)(m0 + col) * DMODEL + hi * 8;
  const _Float16* arow1 = arow0 + 16 * (size_t)DMODEL;
  for (int kk = 0; kk < DMODEL; kk += 32) {
    __builtin_prefetch(arow0 + kk + 128, 0, 0);
    v16h a[2];
    {
      v8h a0 = *(const v8h*)(arow0 + kk);
      v8h a1 = *(const v8h*)(arow0 + kk + 16);
      v8h a2 = *(const v8h*)(arow1 + kk);
      v8h a3 = *(const v8h*)(arow1 + kk + 16);
      #pragma unroll
      for (int i = 0; i < 8; ++i) {
        a[0][i] = a0[i]; a[0][8 + i] = a1[i];
        a[1][i] = a2[i]; a[1][8 + i] = a3[i];
      }
    }
    #pragma unroll
    for (int j = 0; j < 4; ++j) {
      const _Float16* brow =
          woh + (size_t)(n0 + 16 * j + col) * DMODEL + kk + hi * 8;
      v8h b0 = *(const v8h*)brow;
      v8h b1 = *(const v8h*)(brow + 16);
      v16h b;
      #pragma unroll
      for (int i = 0; i < 8; ++i) { b[i] = b0[i]; b[8 + i] = b1[i]; }
      acc[0][j] = wmma16(a[0], b, acc[0][j]);
      acc[1][j] = wmma16(a[1], b, acc[1][j]);
    }
  }
  #pragma unroll
  for (int mi = 0; mi < 2; ++mi)
    #pragma unroll
    for (int j = 0; j < 4; ++j)
      #pragma unroll
      for (int r = 0; r < 8; ++r)
        out[(size_t)(m0 + 16 * mi + r + hi * 8) * DMODEL + n0 + 16 * j + col] =
            acc[mi][j][r] + bo[n0 + 16 * j + col];
}

// ---------------------------------------------------------------------------
extern "C" void kernel_launch(void* const* d_in, const int* in_sizes, int n_in,
                              void* d_out, int out_size, void* d_ws,
                              size_t ws_size, hipStream_t stream) {
  (void)in_sizes; (void)n_in; (void)out_size; (void)ws_size;
  const float* x  = (const float*)d_in[0];
  const float* Wq = (const float*)d_in[1];
  const float* Wk = (const float*)d_in[2];
  const float* Wv = (const float*)d_in[3];
  const float* Wo = (const float*)d_in[4];
  const float* bo = (const float*)d_in[5];
  float* out = (float*)d_out;

  _Float16* xh  = (_Float16*)d_ws;
  _Float16* wqh = xh  + (size_t)NSEQ * DMODEL;
  _Float16* wkh = wqh + (size_t)DMODEL * DMODEL;
  _Float16* wvh = wkh + (size_t)DMODEL * DMODEL;
  _Float16* woh = wvh + (size_t)DMODEL * DMODEL;
  _Float16* qh  = woh + (size_t)DMODEL * DMODEL;
  _Float16* kh  = qh  + (size_t)NSEQ * DMODEL;
  _Float16* vh  = kh  + (size_t)NSEQ * DMODEL;
  _Float16* ch  = vh  + (size_t)NSEQ * DMODEL;

  auto cvt = [&](const float* s, _Float16* d, size_t n) {
    int n4 = (int)(n / 4);
    cvt_f32_f16<<<(n4 + 255) / 256, 256, 0, stream>>>(s, d, n4);
  };
  cvt(x,  xh,  (size_t)NSEQ * DMODEL);
  cvt(Wq, wqh, (size_t)DMODEL * DMODEL);
  cvt(Wk, wkh, (size_t)DMODEL * DMODEL);
  cvt(Wv, wvh, (size_t)DMODEL * DMODEL);
  cvt(Wo, woh, (size_t)DMODEL * DMODEL);

  qkv_proj_kernel<<<dim3(NSEQ / 128, DMODEL / 128, 3), 256, 0, stream>>>(
      xh, wqh, wkh, wvh, qh, kh, vh);
  flash_attn_kernel<<<dim3(NSEQ / 128, NHEAD), 256, 0, stream>>>(qh, kh, vh,
                                                                 ch);
  out_proj_kernel<<<dim3(NSEQ / 128, DMODEL / 128), 256, 0, stream>>>(ch, woh,
                                                                      bo, out);
}
